// GCN_49881750176158
// MI455X (gfx1250) — matrix operand order repaired and checked by reading, not compiled, not run on hardware
//
#include <hip/hip_runtime.h>
#include <hip/hip_bf16.h>

typedef __attribute__((ext_vector_type(16))) _Float16 v16h;
typedef __attribute__((ext_vector_type(8)))  float    v8f;

#define FEAT 128
#define CLASSES 10
#define GRAPHS 64

__device__ __forceinline__ void atomicAddF32(float* p, float v) {
    __hip_atomic_fetch_add(p, v, __ATOMIC_RELAXED, __HIP_MEMORY_SCOPE_AGENT);
}

// ---------------- degree / norm ----------------
__global__ void gcn_deg_init(float* deg, int n) {
    int i = blockIdx.x * blockDim.x + threadIdx.x;
    if (i < n) deg[i] = 1.0f;               // self-loop
}

__global__ void gcn_deg_edges(const int* __restrict__ dst, float* deg, int ne) {
    int e = blockIdx.x * blockDim.x + threadIdx.x;
    if (e < ne) atomicAddF32(&deg[dst[e]], 1.0f);
}

__global__ void gcn_dinv(float* deg, int n) {
    int i = blockIdx.x * blockDim.x + threadIdx.x;
    if (i < n) deg[i] = rsqrtf(deg[i]);     // deg >= 1 always (self-loop)
}

// ---------------- weight f32 -> f16, packed into WMMA B-fragment order ------
// Layout: [col-tile ct 0..7][k-step kt 0..3][lane 0..31][j 0..15] contiguous.
// Lane = (half,m): B 32x16 fragment element j holds K = kt*32 + half*16 + j,
// column N = ct*16 + m. So each lane's v16h is one contiguous 32-byte chunk.
__global__ void gcn_wpack(const float* __restrict__ W, _Float16* __restrict__ Wp) {
    int i = blockIdx.x * blockDim.x + threadIdx.x;
    if (i >= FEAT * FEAT) return;
    int j    = i & 15;
    int lane = (i >> 4) & 31;
    int kt   = (i >> 9) & 3;
    int ct   = i >> 11;
    int half = lane >> 4, m = lane & 15;
    int k    = kt * 32 + half * 16 + j;
    int ncol = ct * 16 + m;
    Wp[i] = (_Float16)W[k * FEAT + ncol];
}

// ---------------- WMMA GEMM: out[N,128] = act[N,128] @ W[128,128] ----------
// Block = 256 threads = 8 waves; block handles 16 rows, wave w cols [16w,16w+16).
// relu_input fuses the previous layer's ReLU into the f32->f16 A-tile staging.
__global__ __launch_bounds__(256) void gcn_gemm_wmma(
    const float* __restrict__ act, const _Float16* __restrict__ Wp,
    float* __restrict__ out, int n, int relu_input)
{
    __shared__ _Float16 As[16][FEAT];       // 4 KB  (A tile, f16)
    __shared__ float    Cs[16][FEAT];       // 8 KB  (C tile, f32, for coalesced store)
    const int tid  = threadIdx.x;
    const int row0 = blockIdx.x * 16;
    const bool full = (row0 + 16) <= n;     // block-uniform tail guard

    // stage A tile (convert + optional ReLU); rows past n -> 0
    for (int idx = tid; idx < 16 * FEAT; idx += 256) {
        int r = idx >> 7, k = idx & 127;
        float v = 0.0f;
        if (full || (row0 + r) < n) {
            v = act[(size_t)(row0 + r) * FEAT + k];
            if (relu_input) v = fmaxf(v, 0.0f);
        }
        As[r][k] = (_Float16)v;
    }
    __syncthreads();

    const int wave = tid >> 5;
    const int lane = tid & 31;
    const int m    = lane & 15;
    const int half = lane >> 4;
    const int n0   = wave * 16;

    v8f c = {0.f, 0.f, 0.f, 0.f, 0.f, 0.f, 0.f, 0.f};
#pragma unroll
    for (int kt = 0; kt < 4; ++kt) {        // K = 128 in 4 steps of 32
        const int kb = kt * 32;
        v16h a;
#pragma unroll
        for (int j = 0; j < 16; ++j) {
            // A 16x32 f16 layout: K = half*16 + j (matches packed B)
            int kA = half * 16 + j;
            a[j] = As[m][kb + kA];          // lowers to ds_load_b128 pairs
        }
        // pre-packed B fragment: one contiguous 32-byte load per lane
        const v16h b = *(const v16h*)(Wp + (((size_t)(wave * 4 + kt) * 32 + lane) << 4));
        c = __builtin_amdgcn_wmma_f32_16x16x32_f16(
            /*neg_a=*/false, a, /*neg_b=*/false, b,
            /*c_mod=*/(short)0, c, /*reuse_a=*/false, /*reuse_b=*/false);
    }

    // D layout: VGPR r -> M = r + 8*half, N = n0 + m; park in LDS, then
    // store the whole 16x128 tile as coalesced float4s (2 per thread).
#pragma unroll
    for (int r = 0; r < 8; ++r)
        Cs[r + 8 * half][n0 + m] = c[r];
    __syncthreads();

    const float4* cs4 = (const float4*)&Cs[0][0];
#pragma unroll
    for (int t = 0; t < 2; ++t) {
        int q    = t * 256 + tid;           // 512 float4s total
        int row  = q >> 5;                  // 32 float4 per row
        int col4 = q & 31;
        if (full || (row0 + row) < n)
            *(float4*)(out + (size_t)(row0 + row) * FEAT + col4 * 4) = cs4[q];
    }
}

// ---------------- aggregation ----------------
// out[i,f] = h[i,f]*dinv[i]^2 + bias[f]   (self-loop + bias)
__global__ void gcn_agg_init(const float* __restrict__ h, const float* __restrict__ dinv,
                             const float* __restrict__ bias, float* __restrict__ out, int n)
{
    int idx = blockIdx.x * blockDim.x + threadIdx.x;
    if (idx < n * FEAT) {
        int i = idx >> 7, f = idx & 127;
        float di = dinv[i];
        out[idx] = h[idx] * di * di + bias[f];
    }
}

// one wave per edge; lane moves a float4 of the 128-wide row; FP32 HW atomics into L2
__global__ __launch_bounds__(256) void gcn_agg_edges(
    const float* __restrict__ h, const float* __restrict__ dinv,
    const int* __restrict__ src, const int* __restrict__ dst,
    float* __restrict__ out, int ne)
{
    int e = blockIdx.x * 8 + (threadIdx.x >> 5);
    if (e >= ne) return;
    int lane = threadIdx.x & 31;
    int s = src[e], d = dst[e];
    float norm = dinv[s] * dinv[d];
    const float4 v = ((const float4*)(h + (size_t)s * FEAT))[lane];
    float* o = out + (size_t)d * FEAT + lane * 4;
    atomicAddF32(o + 0, v.x * norm);
    atomicAddF32(o + 1, v.y * norm);
    atomicAddF32(o + 2, v.z * norm);
    atomicAddF32(o + 3, v.w * norm);
}

// ---------------- pooling + head ----------------
__global__ void gcn_pool_zero(float* emb, float* cnt) {
    int i = blockIdx.x * blockDim.x + threadIdx.x;
    if (i < GRAPHS * FEAT) emb[i] = 0.0f;
    if (i < GRAPHS)        cnt[i] = 0.0f;
}

__global__ __launch_bounds__(128) void gcn_pool_acc(
    const float* __restrict__ h, const int* __restrict__ batch,
    float* emb, float* cnt, int n)
{
    int i = blockIdx.x;
    int f = threadIdx.x;
    if (i >= n) return;
    int g = batch[i];
    atomicAddF32(&emb[g * FEAT + f], h[(size_t)i * FEAT + f]);
    if (f == 0) atomicAddF32(&cnt[g], 1.0f);
}

// logits[g,c] = sum_f emb[g,f]*Wlin[f,c] + blin[c]; also writes emb to d_out
__global__ __launch_bounds__(128) void gcn_head(
    const float* __restrict__ emb_sum, const float* __restrict__ cnt,
    const float* __restrict__ Wlin, const float* __restrict__ blin,
    float* __restrict__ out)
{
    __shared__ float red[FEAT];
    int g = blockIdx.x;
    int f = threadIdx.x;
    float c = cnt[g];
    c = c > 1.0f ? c : 1.0f;
    float e = emb_sum[g * FEAT + f] / c;
    out[GRAPHS * CLASSES + g * FEAT + f] = e;   // emb output (after logits block)
    for (int cls = 0; cls < CLASSES; ++cls) {
        red[f] = e * Wlin[f * CLASSES + cls];
        __syncthreads();
        for (int s = 64; s > 0; s >>= 1) {
            if (f < s) red[f] += red[f + s];
            __syncthreads();
        }
        if (f == 0) out[g * CLASSES + cls] = red[0] + blin[cls];
        __syncthreads();
    }
}

// ---------------- launcher ----------------
static inline size_t alignUp(size_t x) { return (x + 255) & ~(size_t)255; }

extern "C" void kernel_launch(void* const* d_in, const int* in_sizes, int n_in,
                              void* d_out, int out_size, void* d_ws, size_t ws_size,
                              hipStream_t stream) {
    const float* x     = (const float*)d_in[0];
    const int*   edge  = (const int*)d_in[1];
    const int*   batch = (const int*)d_in[2];
    const float* W1 = (const float*)d_in[3]; const float* b1 = (const float*)d_in[4];
    const float* W2 = (const float*)d_in[5]; const float* b2 = (const float*)d_in[6];
    const float* W3 = (const float*)d_in[7]; const float* b3 = (const float*)d_in[8];
    const float* Wlin = (const float*)d_in[9]; const float* blin = (const float*)d_in[10];
    float* out = (float*)d_out;

    const int n  = in_sizes[0] / FEAT;
    const int ne = in_sizes[1] / 2;
    const int* srcp = edge;
    const int* dstp = edge + ne;

    // workspace carve-up
    char* w = (char*)d_ws;
    size_t szH = alignUp((size_t)n * FEAT * sizeof(float));
    float* bufP = (float*)w;               w += szH;
    float* bufQ = (float*)w;               w += szH;
    float* dinv = (float*)w;               w += alignUp((size_t)n * sizeof(float));
    _Float16* Wf16 = (_Float16*)w;         w += alignUp(3 * FEAT * FEAT * sizeof(_Float16));
    float* emb = (float*)w;                w += alignUp(GRAPHS * FEAT * sizeof(float));
    float* cnt = (float*)w;

    const int T = 256;
    // degrees -> dinv
    gcn_deg_init<<<(n + T - 1) / T, T, 0, stream>>>(dinv, n);
    gcn_deg_edges<<<(ne + T - 1) / T, T, 0, stream>>>(dstp, dinv, ne);
    gcn_dinv<<<(n + T - 1) / T, T, 0, stream>>>(dinv, n);

    // weights to f16, pre-packed into WMMA B-fragment order
    const int WCNT = FEAT * FEAT;
    gcn_wpack<<<(WCNT + T - 1) / T, T, 0, stream>>>(W1, Wf16);
    gcn_wpack<<<(WCNT + T - 1) / T, T, 0, stream>>>(W2, Wf16 + WCNT);
    gcn_wpack<<<(WCNT + T - 1) / T, T, 0, stream>>>(W3, Wf16 + 2 * WCNT);

    const int gemmGrid = (n + 15) / 16;
    const int aggGrid  = (n * FEAT + T - 1) / T;
    const int edgeGrid = (ne + 7) / 8;

    // Layer 1: GEMM(x) -> Q ; aggregate Q -> P (bias b1); ReLU deferred into next GEMM
    gcn_gemm_wmma<<<gemmGrid, T, 0, stream>>>(x, Wf16, bufQ, n, 0);
    gcn_agg_init<<<aggGrid, T, 0, stream>>>(bufQ, dinv, b1, bufP, n);
    gcn_agg_edges<<<edgeGrid, T, 0, stream>>>(bufQ, dinv, srcp, dstp, bufP, ne);

    // Layer 2 (ReLU fused into A-staging)
    gcn_gemm_wmma<<<gemmGrid, T, 0, stream>>>(bufP, Wf16 + WCNT, bufQ, n, 1);
    gcn_agg_init<<<aggGrid, T, 0, stream>>>(bufQ, dinv, b2, bufP, n);
    gcn_agg_edges<<<edgeGrid, T, 0, stream>>>(bufQ, dinv, srcp, dstp, bufP, ne);

    // Layer 3 (ReLU fused; no ReLU on output)
    gcn_gemm_wmma<<<gemmGrid, T, 0, stream>>>(bufP, Wf16 + 2 * WCNT, bufQ, n, 1);
    gcn_agg_init<<<aggGrid, T, 0, stream>>>(bufQ, dinv, b3, bufP, n);
    gcn_agg_edges<<<edgeGrid, T, 0, stream>>>(bufQ, dinv, srcp, dstp, bufP, ne);

    // mean pool + linear head
    gcn_pool_zero<<<(GRAPHS * FEAT + T - 1) / T, T, 0, stream>>>(emb, cnt);
    gcn_pool_acc<<<n, FEAT, 0, stream>>>(bufP, batch, emb, cnt, n);
    gcn_head<<<GRAPHS, FEAT, 0, stream>>>(emb, cnt, Wlin, blin, out);
}
